// StructureAwareMLP_38714835206727
// MI455X (gfx1250) — matrix operand order repaired and checked by reading, not compiled, not run on hardware
//
#include <hip/hip_runtime.h>
#include <hip/hip_bf16.h>

// ---------------- problem constants (match reference) ----------------
#define NN      8192
#define EE      262144
#define D_IN    768
#define D_STRUCT 32
#define D_HID   768
#define D_OUTD  768
#define K1      (D_IN + D_STRUCT)   // 800
#define MEM_K   64
#define MEM_D   768
#define NGRAPH  64

typedef __attribute__((ext_vector_type(16))) __bf16 v16bf;
typedef __attribute__((ext_vector_type(8)))  __bf16 v8bf;
typedef __attribute__((ext_vector_type(8)))  float  v8f;

// ---------------- generic WMMA GEMM (templated on N, K) ----------------
// C[M,N] = act( scale * (A_bf16[M,K] @ B_bf16[K,N]) + bias[N] )
// act: 0 = none, 1 = PReLU(alpha = actParam[0]), 2 = tanh
// REQUIRES: M % 64 == 0, K % 32 == 0 (all call sites satisfy this).
// N may be < gridDim.x*BN (masked; N even).
#define BM 64
#define BN 128
#define BK 32

template<int N, int K>
__global__ __launch_bounds__(256)
void k_gemm_bf16(const __bf16* __restrict__ A, const __bf16* __restrict__ B,
                 const float* __restrict__ bias, const float* __restrict__ actParam,
                 float scale, int act, int M,
                 float* __restrict__ outF, __bf16* __restrict__ outB)
{
    // double-buffered LDS: A row-major 64x32, B transposed [n][k] 128x32
    __shared__ __bf16 sA [2][BM * BK];
    __shared__ __bf16 sBt[2][BN * BK];

    const int tid  = threadIdx.x;
    const int lane = tid & 31;
    const int wave = tid >> 5;
    const int wm   = wave & 3;   // 4 wave rows  -> 64 M
    const int wn   = wave >> 2;  // 2 wave cols  -> 2*64 N
    const int bm   = blockIdx.y * BM;
    const int bn   = blockIdx.x * BN;

    // cooperative-fill coordinates
    const int ar  = tid >> 2;          // A row 0..63
    const int ak  = (tid & 3) * 8;     // A k-chunk 0,8,16,24
    const int bn2 = (tid & 63) * 2;    // B col pair 0..126
    const int bk8 = (tid >> 6) * 8;    // B k-chunk 0,8,16,24

    const __bf16* aSrc = A + (size_t)(bm + ar) * K + ak;
    const int  gn  = bn + bn2;
    const bool nOK = (gn < N);                 // gn even, N even
    const __bf16* bSrc = B + (size_t)bk8 * N + gn;

    v8f z = {0.f,0.f,0.f,0.f,0.f,0.f,0.f,0.f};
    v8f acc[4] = {z, z, z, z};

    // fragment coordinates (ISA 7.12.2 bf16 layouts)
    const int mrow = wm * 16 + (lane & 15);
    const int akb  = (lane >> 4) << 3;   // 0 or 8
    const int bkb  = (lane >> 4) << 3;   // 0 or 8

    // staged registers for the next tile
    v8bf aReg, b0Reg, b1Reg;

    auto stage = [&](int k0) {
        aReg = *(const v8bf*)(aSrc + k0);
        if (nOK) {
            #pragma unroll
            for (int kk = 0; kk < 8; ++kk) {
                unsigned u = *(const unsigned*)(bSrc + (size_t)(k0 + kk) * N);
                b0Reg[kk] = ((const __bf16*)&u)[0];
                b1Reg[kk] = ((const __bf16*)&u)[1];
            }
        } else {
            #pragma unroll
            for (int kk = 0; kk < 8; ++kk) { b0Reg[kk] = (__bf16)0.f; b1Reg[kk] = (__bf16)0.f; }
        }
    };
    auto commit = [&](int buf) {
        *(v8bf*)&sA [buf][ar * BK + ak]          = aReg;
        *(v8bf*)&sBt[buf][(bn2)     * BK + bk8]  = b0Reg;
        *(v8bf*)&sBt[buf][(bn2 + 1) * BK + bk8]  = b1Reg;
    };

    // prologue: tile 0 into buffer 0
    stage(0);
    commit(0);
    __syncthreads();

    const int kTiles = K / BK;
    for (int kt = 0; kt < kTiles; ++kt) {
        const int  cur  = kt & 1;
        const bool more = (kt + 1 < kTiles);

        // issue next tile's global loads early (hide under WMMA)
        if (more) stage((kt + 1) * BK);

        // A fragment: two aligned 16B LDS loads
        union { v16bf v; v8bf h[2]; } af;
        af.h[0] = *(const v8bf*)&sA[cur][mrow * BK + akb];
        af.h[1] = *(const v8bf*)&sA[cur][mrow * BK + akb + 16];

        #pragma unroll
        for (int j = 0; j < 4; ++j) {
            const int ncol = wn * 64 + j * 16 + (lane & 15);
            union { v16bf v; v8bf h[2]; } bf;
            bf.h[0] = *(const v8bf*)&sBt[cur][ncol * BK + bkb];
            bf.h[1] = *(const v8bf*)&sBt[cur][ncol * BK + bkb + 16];

            acc[j] = __builtin_amdgcn_wmma_f32_16x16x32_bf16(
                         false, af.v, false, bf.v,
                         (short)0, acc[j], false, false);
        }

        if (more) commit(cur ^ 1);
        __syncthreads();
    }

    const float alpha = (act == 1 && actParam) ? actParam[0] : 0.0f;
    #pragma unroll
    for (int j = 0; j < 4; ++j) {
        const int gcol = bn + wn * 64 + j * 16 + (lane & 15);
        if (gcol >= N) continue;
        const float bv = bias ? bias[gcol] : 0.0f;
        #pragma unroll
        for (int v = 0; v < 8; ++v) {
            const int gm = bm + wm * 16 + v + ((lane >> 4) << 3);
            float x = acc[j][v] * scale + bv;
            if (act == 1)      x = (x > 0.f) ? x : alpha * x;
            else if (act == 2) x = tanhf(x);
            if (outF) outF[(size_t)gm * N + gcol] = x;
            if (outB) outB[(size_t)gm * N + gcol] = (__bf16)x;
        }
    }
}

// ---------------- helper kernels ----------------
__global__ __launch_bounds__(256)
void k_copy_f32(const float* __restrict__ src, float* __restrict__ dst, int n) {
    int i = blockIdx.x * 256 + threadIdx.x;
    if (i < n) dst[i] = src[i];
}

__global__ __launch_bounds__(256)
void k_zero_f32(float* __restrict__ p, int n) {
    int i = blockIdx.x * 256 + threadIdx.x;
    if (i < n) p[i] = 0.0f;
}

__global__ __launch_bounds__(256)
void k_cvt_bf16(const float* __restrict__ src, __bf16* __restrict__ dst, int n) {
    int i = blockIdx.x * 256 + threadIdx.x;
    if (i < n) dst[i] = (__bf16)src[i];
}

// memory[64,768] -> memT_bf16[768,64]
__global__ __launch_bounds__(256)
void k_transpose_cvt(const float* __restrict__ src, __bf16* __restrict__ dst) {
    int i = blockIdx.x * 256 + threadIdx.x;
    if (i < MEM_K * MEM_D) {
        int k = i / MEM_D, d = i % MEM_D;
        dst[d * MEM_K + k] = (__bf16)src[i];
    }
}

// feat[col[e]] += edge_attr[e]   (flattened over E*D_IN)
__global__ __launch_bounds__(256)
void k_edge_scatter(const float* __restrict__ edge_attr,
                    const int* __restrict__ edge_index,
                    float* __restrict__ feat) {
    long i = (long)blockIdx.x * 256 + threadIdx.x;
    if (i < (long)EE * D_IN) {
        int e = (int)(i / D_IN);
        int d = (int)(i % D_IN);
        int col = edge_index[EE + e];   // second row of edge_index
        atomicAdd(&feat[(long)col * D_IN + d], edge_attr[i]);
    }
}

// build Xb[n, 0:768]=bf16(feat); Xb[n, 768+j]=bf16(relu(ppr*Ws+bs))
__global__ __launch_bounds__(256)
void k_build_x(const float* __restrict__ feat, const float* __restrict__ ppr,
               const float* __restrict__ Ws, const float* __restrict__ bs,
               __bf16* __restrict__ Xb) {
    int n = blockIdx.x;
    float p = ppr[n];
    for (int i = threadIdx.x; i < K1; i += 256) {
        float v;
        if (i < D_IN) v = feat[(long)n * D_IN + i];
        else {
            int j = i - D_IN;
            v = p * Ws[j] + bs[j];
            v = v > 0.f ? v : 0.f;
        }
        Xb[(long)n * K1 + i] = (__bf16)v;
    }
}

// segment-sum node_feats into gsum[64,768]
__global__ __launch_bounds__(256)
void k_graph_sum(const float* __restrict__ nf, const int* __restrict__ bid,
                 float* __restrict__ gsum) {
    long i = (long)blockIdx.x * 256 + threadIdx.x;
    if (i < (long)NN * D_OUTD) {
        int n = (int)(i / D_OUTD);
        int d = (int)(i % D_OUTD);
        atomicAdd(&gsum[(long)bid[n] * D_OUTD + d], nf[i]);
    }
}

__global__ __launch_bounds__(256)
void k_graph_cnt(const int* __restrict__ bid, float* __restrict__ gcnt) {
    int n = blockIdx.x * 256 + threadIdx.x;
    if (n < NN) atomicAdd(&gcnt[bid[n]], 1.0f);
}

__global__ __launch_bounds__(256)
void k_graph_mean(const float* __restrict__ gsum, const float* __restrict__ gcnt,
                  __bf16* __restrict__ gf) {
    int i = blockIdx.x * 256 + threadIdx.x;
    if (i < NGRAPH * D_OUTD) {
        float c = gcnt[i / D_OUTD];
        gf[i] = (__bf16)(gsum[i] / fmaxf(c, 1.0f));
    }
}

// per-row softmax over 64 scores; one wave32 per row (2 values/lane)
__global__ __launch_bounds__(256)
void k_softmax(const float* __restrict__ scores, __bf16* __restrict__ wb) {
    int lane = threadIdx.x & 31;
    int wave = threadIdx.x >> 5;
    int row  = blockIdx.x * 8 + wave;
    if (row >= NN) return;
    const float* s = scores + (long)row * MEM_K;
    float v0 = s[lane], v1 = s[lane + 32];
    float m = fmaxf(v0, v1);
    #pragma unroll
    for (int o = 16; o > 0; o >>= 1) m = fmaxf(m, __shfl_xor(m, o, 32));
    float e0 = __expf(v0 - m), e1 = __expf(v1 - m);
    float sum = e0 + e1;
    #pragma unroll
    for (int o = 16; o > 0; o >>= 1) sum += __shfl_xor(sum, o, 32);
    float inv = 1.0f / sum;
    wb[(long)row * MEM_K + lane]      = (__bf16)(e0 * inv);
    wb[(long)row * MEM_K + lane + 32] = (__bf16)(e1 * inv);
}

// loss = mean((a-b)^2); atomic accumulation of pre-scaled block sums
__global__ __launch_bounds__(256)
void k_loss(const float* __restrict__ a, const float* __restrict__ b,
            float* __restrict__ out, long n, float invN) {
    __shared__ float sdata[8];
    float s = 0.f;
    for (long i = (long)blockIdx.x * 256 + threadIdx.x; i < n;
         i += (long)gridDim.x * 256) {
        float d = a[i] - b[i];
        s += d * d;
    }
    #pragma unroll
    for (int o = 16; o > 0; o >>= 1) s += __shfl_xor(s, o, 32);
    if ((threadIdx.x & 31) == 0) sdata[threadIdx.x >> 5] = s;
    __syncthreads();
    if (threadIdx.x == 0) {
        float t = 0.f;
        #pragma unroll
        for (int w = 0; w < 8; ++w) t += sdata[w];
        atomicAdd(out, t * invN);
    }
}

// ---------------- host-side orchestration ----------------
static inline int cdiv(long a, long b) { return (int)((a + b - 1) / b); }

extern "C" void kernel_launch(void* const* d_in, const int* in_sizes, int n_in,
                              void* d_out, int out_size, void* d_ws, size_t ws_size,
                              hipStream_t stream) {
    const float* feature   = (const float*)d_in[0];
    const float* ppr       = (const float*)d_in[1];
    const float* edge_attr = (const float*)d_in[2];
    const float* Ws        = (const float*)d_in[3];
    const float* bs        = (const float*)d_in[4];
    const float* W1        = (const float*)d_in[5];
    const float* b1        = (const float*)d_in[6];
    const float* prelu_a   = (const float*)d_in[7];
    const float* W2        = (const float*)d_in[8];
    const float* b2        = (const float*)d_in[9];
    const float* Wp1       = (const float*)d_in[10];
    const float* bp1       = (const float*)d_in[11];
    const float* Wp2       = (const float*)d_in[12];
    const float* bp2       = (const float*)d_in[13];
    const float* memory    = (const float*)d_in[14];
    const int*   edge_idx  = (const int*)d_in[15];
    const int*   batch_ids = (const int*)d_in[16];

    float* out = (float*)d_out;
    float* out_pooled = out;                              // [64,768]
    float* out_re     = out + (long)NGRAPH * D_OUTD;      // [8192,768]
    float* out_loss   = out_re + (long)NN * D_OUTD;       // [1]

    // ---- workspace layout ----
    char* w = (char*)d_ws;
    size_t off = 0;
    auto alloc = [&](size_t bytes) { void* p = w + off; off = (off + bytes + 255) & ~(size_t)255; return p; };
    float*  feat     = (float*) alloc((size_t)NN * D_IN * 4);
    __bf16* Xb       = (__bf16*)alloc((size_t)NN * K1 * 2);
    __bf16* W1b      = (__bf16*)alloc((size_t)K1 * D_HID * 2);
    __bf16* W2b      = (__bf16*)alloc((size_t)D_HID * D_OUTD * 2);
    __bf16* Wp1b     = (__bf16*)alloc((size_t)D_HID * D_HID * 2);
    __bf16* Wp2b     = (__bf16*)alloc((size_t)D_HID * D_OUTD * 2);
    __bf16* memb     = (__bf16*)alloc((size_t)MEM_K * MEM_D * 2);
    __bf16* memTb    = (__bf16*)alloc((size_t)MEM_D * MEM_K * 2);
    __bf16* h_bf     = (__bf16*)alloc((size_t)NN * D_HID * 2);
    float*  node_f32 = (float*) alloc((size_t)NN * D_OUTD * 4);
    __bf16* node_bf  = (__bf16*)alloc((size_t)NN * D_OUTD * 2);
    float*  gsum     = (float*) alloc((size_t)NGRAPH * D_OUTD * 4);
    float*  gcnt     = (float*) alloc((size_t)NGRAPH * 4);
    __bf16* gf_bf    = (__bf16*)alloc((size_t)NGRAPH * D_OUTD * 2);
    __bf16* t_bf     = (__bf16*)alloc((size_t)NGRAPH * D_HID * 2);
    float*  scores   = (float*) alloc((size_t)NN * MEM_K * 4);
    __bf16* wb       = (__bf16*)alloc((size_t)NN * MEM_K * 2);
    (void)ws_size; (void)n_in; (void)in_sizes; (void)out_size;

    // 1) feat = feature; zero accumulators
    k_copy_f32<<<cdiv((long)NN * D_IN, 256), 256, 0, stream>>>(feature, feat, NN * D_IN);
    k_zero_f32<<<cdiv(NGRAPH * D_OUTD + NGRAPH, 256), 256, 0, stream>>>(gsum, NGRAPH * D_OUTD + NGRAPH);
    k_zero_f32<<<1, 256, 0, stream>>>(out_loss, 1);

    // 2) scatter-add edges (bandwidth dominant: ~0.8GB read)
    k_edge_scatter<<<cdiv((long)EE * D_IN, 256), 256, 0, stream>>>(edge_attr, edge_idx, feat);

    // 3) build bf16 input matrix [N, 800] = [feat | relu(ppr*Ws+bs)]
    k_build_x<<<NN, 256, 0, stream>>>(feat, ppr, Ws, bs, Xb);

    // 4) convert weights / memory to bf16
    k_cvt_bf16<<<cdiv(K1 * D_HID, 256), 256, 0, stream>>>(W1, W1b, K1 * D_HID);
    k_cvt_bf16<<<cdiv(D_HID * D_OUTD, 256), 256, 0, stream>>>(W2, W2b, D_HID * D_OUTD);
    k_cvt_bf16<<<cdiv(D_HID * D_HID, 256), 256, 0, stream>>>(Wp1, Wp1b, D_HID * D_HID);
    k_cvt_bf16<<<cdiv(D_HID * D_OUTD, 256), 256, 0, stream>>>(Wp2, Wp2b, D_HID * D_OUTD);
    k_cvt_bf16<<<cdiv(MEM_K * MEM_D, 256), 256, 0, stream>>>(memory, memb, MEM_K * MEM_D);
    k_transpose_cvt<<<cdiv(MEM_K * MEM_D, 256), 256, 0, stream>>>(memory, memTb);

    // 5) h = PReLU(X @ W1 + b1)            [8192,800]x[800,768]
    {
        dim3 g(cdiv(D_HID, BN), cdiv(NN, BM));
        k_gemm_bf16<D_HID, K1><<<g, 256, 0, stream>>>(Xb, W1b, b1, prelu_a, 1.0f, 1,
                                                      NN, nullptr, h_bf);
    }
    // 6) node_feats = h @ W2 + b2          [8192,768]x[768,768]
    {
        dim3 g(cdiv(D_OUTD, BN), cdiv(NN, BM));
        k_gemm_bf16<D_OUTD, D_HID><<<g, 256, 0, stream>>>(h_bf, W2b, b2, nullptr, 1.0f, 0,
                                                          NN, node_f32, node_bf);
    }
    // 7) graph mean pooling
    k_graph_sum<<<cdiv((long)NN * D_OUTD, 256), 256, 0, stream>>>(node_f32, batch_ids, gsum);
    k_graph_cnt<<<cdiv(NN, 256), 256, 0, stream>>>(batch_ids, gcnt);
    k_graph_mean<<<cdiv(NGRAPH * D_OUTD, 256), 256, 0, stream>>>(gsum, gcnt, gf_bf);

    // 8) pooled = tanh(gf @ Wp1 + bp1) @ Wp2 + bp2    [64,768] chains
    {
        dim3 g(cdiv(D_HID, BN), cdiv(NGRAPH, BM));
        k_gemm_bf16<D_HID, D_OUTD><<<g, 256, 0, stream>>>(gf_bf, Wp1b, bp1, nullptr, 1.0f, 2,
                                                          NGRAPH, nullptr, t_bf);
        k_gemm_bf16<D_OUTD, D_HID><<<g, 256, 0, stream>>>(t_bf, Wp2b, bp2, nullptr, 1.0f, 0,
                                                          NGRAPH, out_pooled, nullptr);
    }
    // 9) scores = node_feats @ memory^T     [8192,768]x[768,64]
    {
        dim3 g(cdiv(MEM_K, BN), cdiv(NN, BM));
        k_gemm_bf16<MEM_K, D_OUTD><<<g, 256, 0, stream>>>(node_bf, memTb, nullptr, nullptr, 1.0f, 0,
                                                          NN, scores, nullptr);
    }
    // 10) row softmax -> weights bf16
    k_softmax<<<cdiv(NN, 8), 256, 0, stream>>>(scores, wb);

    // 11) re = (w @ memory) / 64            [8192,64]x[64,768]
    {
        dim3 g(cdiv(MEM_D, BN), cdiv(NN, BM));
        k_gemm_bf16<MEM_D, MEM_K><<<g, 256, 0, stream>>>(wb, memb, nullptr, nullptr,
                                                         1.0f / (float)MEM_K, 0,
                                                         NN, out_re, nullptr);
    }
    // 12) loss = mean((node_feats - re)^2)
    k_loss<<<4096, 256, 0, stream>>>(node_f32, out_re, out_loss,
                                     (long)NN * D_OUTD, 1.0f / ((float)NN * (float)D_OUTD));
}
